// TransformerBlockWithRoPE_57836029608334
// MI455X (gfx1250) — compile-verified, matrix operand order checked by reading, not run on hardware
//
#include <hip/hip_runtime.h>

#define DM   1024
#define TSEQ 2048
#define BSZ  2
#define NH   16
#define HD   64
#define DFF  4096
#define MROWS (BSZ*TSEQ)   // 4096

typedef __attribute__((ext_vector_type(16))) __bf16 v16bf;
typedef __attribute__((ext_vector_type(8)))  float  v8f;

struct alignas(16) V4 { unsigned a, b, c, d; };
union FragA { v16bf v; unsigned short u[16]; V4 q[2]; };
union FragC { v8f v; float f[8]; };

__device__ __forceinline__ unsigned short bf16u(float f) {
  union { float f; unsigned u; } x; x.f = f;
  unsigned r = x.u + 0x7FFFu + ((x.u >> 16) & 1u);
  return (unsigned short)(r >> 16);
}

// ---- CDNA5 async global->LDS copy (ASYNCcnt), guarded with fallback ----
#if __has_builtin(__builtin_amdgcn_global_load_async_to_lds_b128)
#define USE_ASYNC 1
#else
#define USE_ASYNC 0
#endif

#if USE_ASYNC
typedef int vsi4 __attribute__((vector_size(4 * sizeof(int))));
typedef __attribute__((address_space(1))) vsi4 g_vsi4;
typedef __attribute__((address_space(3))) vsi4 l_vsi4;
#endif

__device__ __forceinline__ void async_copy16(const unsigned short* g, unsigned short* l) {
#if USE_ASYNC
  __builtin_amdgcn_global_load_async_to_lds_b128(
      (g_vsi4*)(const_cast<unsigned short*>(g)),
      (l_vsi4*)(l), 0, 0);
#else
  *reinterpret_cast<V4*>(l) = *reinterpret_cast<const V4*>(g);
#endif
}

__device__ __forceinline__ void wait_async() {
#if USE_ASYNC
#if __has_builtin(__builtin_amdgcn_s_wait_asynccnt)
  __builtin_amdgcn_s_wait_asynccnt(0);
#else
  asm volatile("s_wait_asynccnt 0x0" ::: "memory");
#endif
#endif
}

// ---------------- elementwise f32 -> bf16 convert ----------------
__global__ __launch_bounds__(256) void cvt_bf16_kernel(const float* __restrict__ src,
                                                       unsigned short* __restrict__ dst, int n) {
  int i = blockIdx.x * 256 + threadIdx.x;
  if (i < n) dst[i] = bf16u(src[i]);
}

// ---------------- LayerNorm: one block per row of 1024 ----------------
__global__ __launch_bounds__(256) void ln_kernel(const float* __restrict__ x,
                                                 const float* __restrict__ g,
                                                 const float* __restrict__ be,
                                                 unsigned short* __restrict__ out) {
  __shared__ float red[16];
  const int row = blockIdx.x;
  const float* xr = x + (size_t)row * DM;
  float vals[4];
  float s = 0.f, ss = 0.f;
#pragma unroll
  for (int i = 0; i < 4; i++) {
    float v = xr[threadIdx.x + i * 256];
    vals[i] = v; s += v; ss += v * v;
  }
#pragma unroll
  for (int off = 16; off >= 1; off >>= 1) {
    s  += __shfl_xor(s,  off, 32);
    ss += __shfl_xor(ss, off, 32);
  }
  const int wave = threadIdx.x >> 5;
  if ((threadIdx.x & 31) == 0) { red[wave] = s; red[8 + wave] = ss; }
  __syncthreads();
  float ts = 0.f, tss = 0.f;
#pragma unroll
  for (int i = 0; i < 8; i++) { ts += red[i]; tss += red[8 + i]; }
  const float mean = ts * (1.f / DM);
  const float var  = tss * (1.f / DM) - mean * mean;
  const float inv  = rsqrtf(var + 1e-5f);
#pragma unroll
  for (int i = 0; i < 4; i++) {
    int c = threadIdx.x + i * 256;
    float h = (vals[i] - mean) * inv * g[c] + be[c];
    out[(size_t)row * DM + c] = bf16u(h);
  }
}

// ---------------- bf16 WMMA GEMM, double-buffered LDS (1 barrier / K-chunk) ----------------
#define LDSB 40   // padded ushort stride: 80B rows -> 16B aligned, conflict-light

template <bool RELU, bool RESID, bool OUTBF16>
__global__ __launch_bounds__(256) void gemm_bf16(const unsigned short* __restrict__ A,
                                                 const unsigned short* __restrict__ Bm,
                                                 const float* __restrict__ bias,
                                                 const float* __restrict__ resid,
                                                 float* __restrict__ outF,
                                                 unsigned short* __restrict__ outH,
                                                 int N, int K) {
  __shared__ unsigned short bt[2][128 * LDSB];   // 20 KB double buffer
  const int tid  = threadIdx.x;
  const int wave = tid >> 5, lane = tid & 31;
  const int half = lane >> 4, l16 = lane & 15;
  const int n0 = blockIdx.x * 128;
  const int m0 = blockIdx.y * 128 + wave * 16;

  FragC c[8];
#pragma unroll
  for (int i = 0; i < 8; i++)
#pragma unroll
    for (int r = 0; r < 8; r++) c[i].f[r] = 0.f;

  const unsigned short* Arow = A + (size_t)(m0 + l16) * K;

  // staging geometry: 512 chunks of 8 bf16 covering the 32x128 B tile
  const int row0 = tid >> 4,         col0 = (tid & 15) * 8;
  const int row1 = (tid + 256) >> 4, col1 = (tid & 15) * 8;  // rows 16..31

  // prologue: fetch + scatter tile 0 into buffer 0
  V4 g0 = *reinterpret_cast<const V4*>(Bm + (size_t)row0 * N + n0 + col0);
  V4 g1 = *reinterpret_cast<const V4*>(Bm + (size_t)row1 * N + n0 + col1);
  {
    const unsigned short* gs0 = reinterpret_cast<const unsigned short*>(&g0);
    const unsigned short* gs1 = reinterpret_cast<const unsigned short*>(&g1);
#pragma unroll
    for (int j = 0; j < 8; j++) bt[0][(col0 + j) * LDSB + row0] = gs0[j];
#pragma unroll
    for (int j = 0; j < 8; j++) bt[0][(col1 + j) * LDSB + row1] = gs1[j];
  }

  const int nIter = K >> 5;
  for (int it = 0; it < nIter; it++) {
    const int cur = it & 1;
    const int k0 = it * 32;
    __syncthreads();   // bt[cur] complete (compiler inserts dscnt wait before barrier)

    // next tile's global loads fly during the WMMA phase
    if (it + 1 < nIter) {
      g0 = *reinterpret_cast<const V4*>(Bm + (size_t)(k0 + 32 + row0) * N + n0 + col0);
      g1 = *reinterpret_cast<const V4*>(Bm + (size_t)(k0 + 32 + row1) * N + n0 + col1);
    }

    FragA a;
    a.q[0] = *reinterpret_cast<const V4*>(Arow + k0 + half * 8);
    a.q[1] = *reinterpret_cast<const V4*>(Arow + k0 + 16 + half * 8);

#pragma unroll
    for (int nt = 0; nt < 8; nt++) {
      FragA b;
      const unsigned short* bp = &bt[cur][(nt * 16 + l16) * LDSB];
      b.q[0] = *reinterpret_cast<const V4*>(bp + half * 8);
      b.q[1] = *reinterpret_cast<const V4*>(bp + 16 + half * 8);
      c[nt].v = __builtin_amdgcn_wmma_f32_16x16x32_bf16(false, a.v, false, b.v,
                                                        (short)0, c[nt].v, false, false);
    }

    // scatter next tile into the other buffer (protected by next iteration's barrier)
    if (it + 1 < nIter) {
      const unsigned short* gs0 = reinterpret_cast<const unsigned short*>(&g0);
      const unsigned short* gs1 = reinterpret_cast<const unsigned short*>(&g1);
#pragma unroll
      for (int j = 0; j < 8; j++) bt[1 - cur][(col0 + j) * LDSB + row0] = gs0[j];
#pragma unroll
      for (int j = 0; j < 8; j++) bt[1 - cur][(col1 + j) * LDSB + row1] = gs1[j];
    }
  }

#pragma unroll
  for (int nt = 0; nt < 8; nt++) {
    int n = n0 + nt * 16 + l16;
    float bv = bias[n];
#pragma unroll
    for (int r = 0; r < 8; r++) {
      int m = m0 + r + half * 8;
      float v = c[nt].f[r] + bv;
      if (RESID) v += resid[(size_t)m * N + n];
      if (RELU)  v = fmaxf(v, 0.f);
      if (OUTBF16) outH[(size_t)m * N + n] = bf16u(v);
      else         outF[(size_t)m * N + n] = v;
    }
  }
}

// ---------------- RoPE: q,k f32 [B,T,C] -> bf16 [B,H,T,64]; Q pre-scaled by 1/sqrt(D) ----------------
__global__ __launch_bounds__(256) void rope_kernel(const float* __restrict__ qf,
                                                   const float* __restrict__ kf,
                                                   unsigned short* __restrict__ qh,
                                                   unsigned short* __restrict__ kh) {
  int idx = blockIdx.x * 256 + threadIdx.x;       // B*H*T*32
  int d = idx & 31;
  int t = (idx >> 5) & (TSEQ - 1);
  int h = (idx >> 16) & 15;
  int b = (idx >> 20) & 1;
  float inv = __expf(-(float)d * (9.210340371976184f / 32.f));  // 10000^(-d/32)
  float ang = (float)t * inv;
  float cs = __cosf(ang), sn = __sinf(ang);
  size_t src = ((size_t)(b * TSEQ + t)) * DM + h * HD + d;
  float x1q = qf[src], x2q = qf[src + 32];
  float x1k = kf[src], x2k = kf[src + 32];
  size_t dst = ((size_t)((b * NH + h) * TSEQ + t)) * HD + d;
  qh[dst]      = bf16u((x1q * cs - x2q * sn) * 0.125f);   // fold 1/sqrt(64) into Q
  qh[dst + 32] = bf16u((x1q * sn + x2q * cs) * 0.125f);
  kh[dst]      = bf16u(x1k * cs - x2k * sn);
  kh[dst + 32] = bf16u(x1k * sn + x2k * cs);
}

__global__ __launch_bounds__(256) void vcvt_kernel(const float* __restrict__ vf,
                                                   unsigned short* __restrict__ vh) {
  int idx = blockIdx.x * 256 + threadIdx.x;       // B*H*T*64
  int d = idx & 63;
  int t = (idx >> 6) & (TSEQ - 1);
  int h = (idx >> 17) & 15;
  int b = (idx >> 21) & 1;
  vh[((size_t)((b * NH + h) * TSEQ + t)) * HD + d] =
      bf16u(vf[((size_t)(b * TSEQ + t)) * DM + h * HD + d]);
}

// ---------------- Flash attention: double-buffered block-cooperative K/V staging ----------------
__global__ __launch_bounds__(256) void attn_kernel(const unsigned short* __restrict__ qh,
                                                   const unsigned short* __restrict__ kh,
                                                   const unsigned short* __restrict__ vh,
                                                   unsigned short* __restrict__ obf) {
  __shared__ unsigned short Kt[2][32 * HD];        // 2x4 KB  row-major [key][d]
  __shared__ unsigned short Vt[2][HD * LDSB];      // 2x5 KB  transposed [d][key]
  __shared__ unsigned short Pt[8 * 16 * LDSB];     // 10 KB per-wave P tiles
  const int tid  = threadIdx.x;
  const int wave = tid >> 5, lane = tid & 31;
  const int half = lane >> 4, l16 = lane & 15;
  const int t0b = blockIdx.x * 128;
  const int t0 = t0b + wave * 16;
  const int h = blockIdx.y, b = blockIdx.z;
  const int bh = b * NH + h;
  const unsigned short* qbase = qh + (size_t)bh * TSEQ * HD;
  const unsigned short* kbase = kh + (size_t)bh * TSEQ * HD;
  const unsigned short* vbase = vh + (size_t)bh * TSEQ * HD;
  unsigned short* pw = &Pt[wave * 16 * LDSB];

  const int srow = tid >> 3;        // staging: key row 0..31
  const int scol = (tid & 7) * 8;   // staging: d column 0..56

  FragA qa0, qa1;
  {
    const unsigned short* qr = qbase + (size_t)(t0 + l16) * HD;
    qa0.q[0] = *reinterpret_cast<const V4*>(qr + half * 8);
    qa0.q[1] = *reinterpret_cast<const V4*>(qr + 16 + half * 8);
    qa1.q[0] = *reinterpret_cast<const V4*>(qr + 32 + half * 8);
    qa1.q[1] = *reinterpret_cast<const V4*>(qr + 48 + half * 8);
  }

  float mrow[8], lrow[8];
  FragC o[4];
#pragma unroll
  for (int r = 0; r < 8; r++) { mrow[r] = -3.0e38f; lrow[r] = 0.f; }
#pragma unroll
  for (int nt = 0; nt < 4; nt++)
#pragma unroll
    for (int r = 0; r < 8; r++) o[nt].f[r] = 0.f;

  // prologue: stage chunk 0 into buffer 0
  async_copy16(kbase + (size_t)srow * HD + scol, &Kt[0][srow * HD + scol]);
  {
    V4 gv = *reinterpret_cast<const V4*>(vbase + (size_t)srow * HD + scol);
    const unsigned short* gs = reinterpret_cast<const unsigned short*>(&gv);
#pragma unroll
    for (int j = 0; j < 8; j++) Vt[0][(scol + j) * LDSB + srow] = gs[j];
  }

  const int nChunks = (t0b + 128) >> 5;
  for (int it = 0; it < nChunks; it++) {
    const int s0 = it * 32;
    const int cur = it & 1;
    wait_async();
    __syncthreads();   // buffers [cur] complete

    // issue next chunk's K async copy + V global load; they fly under the WMMAs
    V4 gv;
    const bool hasNext = (it + 1 < nChunks);
    if (hasNext) {
      const int sn = s0 + 32;
      async_copy16(kbase + (size_t)(sn + srow) * HD + scol, &Kt[1 - cur][srow * HD + scol]);
      gv = *reinterpret_cast<const V4*>(vbase + (size_t)(sn + srow) * HD + scol);
    }

    if (s0 < t0 + 16) {   // wave-uniform guard (EXEC stays all-ones for WMMA)
      // batch-load all four K fragments, then run the four QK^T WMMAs
      FragA kb00, kb01, kb10, kb11;
      {
        const unsigned short* kr0 = &Kt[cur][l16 * HD];          // keys s0..s0+15
        const unsigned short* kr1 = &Kt[cur][(16 + l16) * HD];   // keys s0+16..s0+31
        kb00.q[0] = *reinterpret_cast<const V4*>(kr0 + half * 8);
        kb00.q[1] = *reinterpret_cast<const V4*>(kr0 + 16 + half * 8);
        kb01.q[0] = *reinterpret_cast<const V4*>(kr0 + 32 + half * 8);
        kb01.q[1] = *reinterpret_cast<const V4*>(kr0 + 48 + half * 8);
        kb10.q[0] = *reinterpret_cast<const V4*>(kr1 + half * 8);
        kb10.q[1] = *reinterpret_cast<const V4*>(kr1 + 16 + half * 8);
        kb11.q[0] = *reinterpret_cast<const V4*>(kr1 + 32 + half * 8);
        kb11.q[1] = *reinterpret_cast<const V4*>(kr1 + 48 + half * 8);
      }
      FragC sc[2];
      {
        FragC z0, z1;
#pragma unroll
        for (int r = 0; r < 8; r++) { z0.f[r] = 0.f; z1.f[r] = 0.f; }
        z0.v = __builtin_amdgcn_wmma_f32_16x16x32_bf16(false, qa0.v, false, kb00.v,
                                                       (short)0, z0.v, false, false);
        z1.v = __builtin_amdgcn_wmma_f32_16x16x32_bf16(false, qa0.v, false, kb10.v,
                                                       (short)0, z1.v, false, false);
        z0.v = __builtin_amdgcn_wmma_f32_16x16x32_bf16(false, qa1.v, false, kb01.v,
                                                       (short)0, z0.v, false, false);
        z1.v = __builtin_amdgcn_wmma_f32_16x16x32_bf16(false, qa1.v, false, kb11.v,
                                                       (short)0, z1.v, false, false);
        sc[0] = z0; sc[1] = z1;
      }
      // causal mask only on diagonal-straddling chunks (Q already carries 1/sqrt(D))
      if (s0 + 31 > t0) {
#pragma unroll
        for (int sub = 0; sub < 2; sub++) {
          int s = s0 + sub * 16 + l16;
#pragma unroll
          for (int r = 0; r < 8; r++) {
            int t = t0 + r + half * 8;
            sc[sub].f[r] = (s > t) ? (sc[sub].f[r] - 1.0e9f) : sc[sub].f[r];
          }
        }
      }
      // online softmax
      float scl[8];
#pragma unroll
      for (int r = 0; r < 8; r++) {
        float rm = fmaxf(sc[0].f[r], sc[1].f[r]);
#pragma unroll
        for (int off = 8; off >= 1; off >>= 1) rm = fmaxf(rm, __shfl_xor(rm, off, 16));
        float mnew = fmaxf(mrow[r], rm);
        scl[r] = __expf(mrow[r] - mnew);
        mrow[r] = mnew;
      }
#pragma unroll
      for (int r = 0; r < 8; r++) {
        float p0 = __expf(sc[0].f[r] - mrow[r]);
        float p1 = __expf(sc[1].f[r] - mrow[r]);
        sc[0].f[r] = p0; sc[1].f[r] = p1;
        float rs = p0 + p1;
#pragma unroll
        for (int off = 8; off >= 1; off >>= 1) rs += __shfl_xor(rs, off, 16);
        lrow[r] = lrow[r] * scl[r] + rs;
#pragma unroll
        for (int nt = 0; nt < 4; nt++) o[nt].f[r] *= scl[r];
      }
      // C-layout -> A-layout transpose of P through wave-private LDS
#pragma unroll
      for (int sub = 0; sub < 2; sub++) {
        int col = sub * 16 + l16;
#pragma unroll
        for (int r = 0; r < 8; r++)
          pw[(r + half * 8) * LDSB + col] = bf16u(sc[sub].f[r]);
      }
      asm volatile("s_wait_dscnt 0x0" ::: "memory");  // cross-lane LDS fence (wave lockstep)
      FragA pa;
      pa.q[0] = *reinterpret_cast<const V4*>(&pw[l16 * LDSB + half * 8]);
      pa.q[1] = *reinterpret_cast<const V4*>(&pw[l16 * LDSB + 16 + half * 8]);
      // accumulate P @ V from transposed LDS tile
#pragma unroll
      for (int nt = 0; nt < 4; nt++) {
        FragA vb;
        const unsigned short* vp = &Vt[cur][(nt * 16 + l16) * LDSB];
        vb.q[0] = *reinterpret_cast<const V4*>(vp + half * 8);
        vb.q[1] = *reinterpret_cast<const V4*>(vp + 16 + half * 8);
        o[nt].v = __builtin_amdgcn_wmma_f32_16x16x32_bf16(false, pa.v, false, vb.v,
                                                          (short)0, o[nt].v, false, false);
      }
    }

    // scatter next V tile into the other buffer (protected by next iteration's barrier)
    if (hasNext) {
      const unsigned short* gs = reinterpret_cast<const unsigned short*>(&gv);
#pragma unroll
      for (int j = 0; j < 8; j++) Vt[1 - cur][(scol + j) * LDSB + srow] = gs[j];
    }
  }
  // finalize and merge heads: obf[B,T,C]
#pragma unroll
  for (int r = 0; r < 8; r++) {
    float inv = 1.f / lrow[r];
    int t = t0 + r + half * 8;
#pragma unroll
    for (int nt = 0; nt < 4; nt++)
      obf[((size_t)(b * TSEQ + t)) * DM + h * HD + nt * 16 + l16] = bf16u(o[nt].f[r] * inv);
  }
}

// ---------------- host side ----------------
extern "C" void kernel_launch(void* const* d_in, const int* in_sizes, int n_in,
                              void* d_out, int out_size, void* d_ws, size_t ws_size,
                              hipStream_t stream) {
  const float* x    = (const float*)d_in[0];
  const float* Wq   = (const float*)d_in[2];
  const float* bq   = (const float*)d_in[3];
  const float* Wk   = (const float*)d_in[4];
  const float* bk   = (const float*)d_in[5];
  const float* Wv   = (const float*)d_in[6];
  const float* bv   = (const float*)d_in[7];
  const float* Wo   = (const float*)d_in[8];
  const float* bo   = (const float*)d_in[9];
  const float* W1   = (const float*)d_in[10];
  const float* b1   = (const float*)d_in[11];
  const float* W2   = (const float*)d_in[12];
  const float* b2   = (const float*)d_in[13];
  const float* g1   = (const float*)d_in[14];
  const float* be1  = (const float*)d_in[15];
  const float* g2   = (const float*)d_in[16];
  const float* be2  = (const float*)d_in[17];
  float* out = (float*)d_out;

  char* ws = (char*)d_ws;
  const size_t MB = 1ull << 20;
  unsigned short* wWq = (unsigned short*)(ws + 0 * MB);   // 2 MB
  unsigned short* wWk = (unsigned short*)(ws + 2 * MB);   // 2 MB
  unsigned short* wWv = (unsigned short*)(ws + 4 * MB);   // 2 MB
  unsigned short* wWo = (unsigned short*)(ws + 6 * MB);   // 2 MB
  unsigned short* wW1 = (unsigned short*)(ws + 8 * MB);   // 8 MB
  unsigned short* wW2 = (unsigned short*)(ws + 16 * MB);  // 8 MB
  unsigned short* hln = (unsigned short*)(ws + 24 * MB);  // 8 MB
  float* qf  = (float*)(ws + 32 * MB);                    // 16 MB (later reused as x1)
  float* kf  = (float*)(ws + 48 * MB);                    // 16 MB (later reused as h2)
  float* vf  = (float*)(ws + 64 * MB);                    // 16 MB (later reused by ffn1)
  unsigned short* qhb = (unsigned short*)(ws + 80 * MB);  // 8 MB
  unsigned short* khb = (unsigned short*)(ws + 88 * MB);  // 8 MB
  unsigned short* vhb = (unsigned short*)(ws + 96 * MB);  // 8 MB
  unsigned short* obf = (unsigned short*)(ws + 104 * MB); // 8 MB
  float* x1          = (float*)(ws + 32 * MB);            // reuse qf
  unsigned short* h2 = (unsigned short*)(ws + 48 * MB);   // reuse kf
  unsigned short* f1 = (unsigned short*)(ws + 64 * MB);   // 32 MB, reuse vf/qhb/khb

  const int nW = DM * DM;       // 1M
  const int nF = DM * DFF;      // 4M

  // 1) weights -> bf16 (L2-resident afterwards)
  cvt_bf16_kernel<<<nW / 256, 256, 0, stream>>>(Wq, wWq, nW);
  cvt_bf16_kernel<<<nW / 256, 256, 0, stream>>>(Wk, wWk, nW);
  cvt_bf16_kernel<<<nW / 256, 256, 0, stream>>>(Wv, wWv, nW);
  cvt_bf16_kernel<<<nW / 256, 256, 0, stream>>>(Wo, wWo, nW);
  cvt_bf16_kernel<<<nF / 256, 256, 0, stream>>>(W1, wW1, nF);
  cvt_bf16_kernel<<<nF / 256, 256, 0, stream>>>(W2, wW2, nF);

  // 2) LN1
  ln_kernel<<<MROWS, 256, 0, stream>>>(x, g1, be1, hln);

  // 3) Q/K/V projections (f32 out)
  dim3 gProj(DM / 128, MROWS / 128);
  gemm_bf16<false, false, false><<<gProj, 256, 0, stream>>>(hln, wWq, bq, nullptr, qf, nullptr, DM, DM);
  gemm_bf16<false, false, false><<<gProj, 256, 0, stream>>>(hln, wWk, bk, nullptr, kf, nullptr, DM, DM);
  gemm_bf16<false, false, false><<<gProj, 256, 0, stream>>>(hln, wWv, bv, nullptr, vf, nullptr, DM, DM);

  // 4) RoPE + layout to [B,H,T,64] bf16
  rope_kernel<<<(BSZ * NH * TSEQ * 32) / 256, 256, 0, stream>>>(qf, kf, qhb, khb);
  vcvt_kernel<<<(BSZ * NH * TSEQ * HD) / 256, 256, 0, stream>>>(vf, vhb);

  // 5) causal flash attention
  attn_kernel<<<dim3(TSEQ / 128, NH, BSZ), 256, 0, stream>>>(qhb, khb, vhb, obf);

  // 6) output projection + residual -> x1
  gemm_bf16<false, true, false><<<gProj, 256, 0, stream>>>(obf, wWo, bo, x, x1, nullptr, DM, DM);

  // 7) LN2
  ln_kernel<<<MROWS, 256, 0, stream>>>(x1, g2, be2, h2);

  // 8) FFN up + ReLU (bf16 out)
  gemm_bf16<true, false, true><<<dim3(DFF / 128, MROWS / 128), 256, 0, stream>>>(
      h2, wW1, b1, nullptr, nullptr, f1, DFF, DM);

  // 9) FFN down + residual -> d_out
  gemm_bf16<false, true, false><<<gProj, 256, 0, stream>>>(f1, wW2, b2, x1, out, nullptr, DM, DFF);

  (void)in_sizes; (void)n_in; (void)out_size; (void)ws_size;
}
